// Attention_3633542332637
// MI455X (gfx1250) — matrix operand order repaired and checked by reading, not compiled, not run on hardware
//
#include <hip/hip_runtime.h>

// ---------------------------------------------------------------------------
// Cosine-similarity linear attention for MI455X (gfx1250, wave32, WMMA).
// Pipeline:
//   cvt (f32->bf16 once for x and weights)
//   qkv proj (bf16 WMMA, async-to-LDS double-buffered)
//   denom -> normalize/scale (wave32 shfl reductions)
//   kv = k^T v (bf16 WMMA, transpose-staged LDS)
//   attn = q kv (bf16 WMMA)
//   out proj (bf16 WMMA, async-to-LDS double-buffered, f32 out)
// All accumulation f32; bf16 only as matrix-core operand format.
// ---------------------------------------------------------------------------

typedef __attribute__((ext_vector_type(16))) __bf16 v16bf;
typedef __attribute__((ext_vector_type(8)))  __bf16 v8bf;
typedef __attribute__((ext_vector_type(4)))  __bf16 v4bf;
typedef __attribute__((ext_vector_type(8)))  float  v8f;
typedef __attribute__((ext_vector_type(4)))  int    v4i;

#define NB   4
#define SB   4096
#define DB   1024
#define HB   16
#define HDB  64

// --- CDNA5 async global->LDS path (probe via __has_builtin, sync fallback) --
#if defined(__gfx1250__) && \
    __has_builtin(__builtin_amdgcn_global_load_async_to_lds_b128) && \
    __has_builtin(__builtin_amdgcn_s_wait_asynccnt)
#define HAS_ASYNC 1
typedef __attribute__((address_space(1))) v4i gas_v4i;   // global v4i
typedef __attribute__((address_space(3))) v4i las_v4i;   // LDS v4i
#else
#define HAS_ASYNC 0
#endif

__device__ __forceinline__ void copy16_g2l(const __bf16* g, __bf16* l) {
#if HAS_ASYNC
  __builtin_amdgcn_global_load_async_to_lds_b128((gas_v4i*)g, (las_v4i*)l, 0, 0);
#else
  *(v8bf*)l = *(const v8bf*)g;
#endif
}
__device__ __forceinline__ void async_wait6() {
#if HAS_ASYNC
  __builtin_amdgcn_s_wait_asynccnt(6);
#endif
}
__device__ __forceinline__ void async_wait0() {
#if HAS_ASYNC
  __builtin_amdgcn_s_wait_asynccnt(0);
#endif
}

__device__ __forceinline__ __bf16 f2bf(float f) {
  unsigned u = __builtin_bit_cast(unsigned, f);
  u += 0x7FFFu + ((u >> 16) & 1u);          // round-to-nearest-even
  return __builtin_bit_cast(__bf16, (unsigned short)(u >> 16));
}
__device__ __forceinline__ float bf2f(__bf16 b) {
  unsigned u = (unsigned)__builtin_bit_cast(unsigned short, b) << 16;
  return __builtin_bit_cast(float, u);
}

__device__ __forceinline__ v8f wmma_bf16(v16bf a, v16bf b, v8f c) {
  // (neg_a, A, neg_b, B, c_mod, C, reuse_a, reuse_b)
  return __builtin_amdgcn_wmma_f32_16x16x32_bf16(false, a, false, b, (short)0, c,
                                                 false, false);
}

// A fragment (16x32 bf16): lane<16 -> K{kb..kb+7, kb+16..kb+23} with kb=ks,
// lane>=16 -> kb=ks+8. Row-contiguous LDS layout -> two 16B vector loads.
__device__ __forceinline__ v16bf ldfragA(const __bf16* row, int ks, int lhi) {
  const int kb = ks + lhi * 8;
  v8bf lo = *(const v8bf*)(row + kb);
  v8bf hi = *(const v8bf*)(row + kb + 16);
  v16bf r;
#pragma unroll
  for (int j = 0; j < 8; ++j) { r[j] = lo[j]; r[j + 8] = hi[j]; }
  return r;
}

// B fragment (32x16 bf16): lane<16 -> K{kb..kb+7, kb+16..kb+23} with kb=ks,
// lane>=16 -> kb=ks+8; B stored column-major in LDS ([n][k]).
__device__ __forceinline__ v16bf ldfragB(const __bf16* row, int ks, int lhi) {
  const int kb = ks + lhi * 8;
  v8bf lo = *(const v8bf*)(row + kb);
  v8bf hi = *(const v8bf*)(row + kb + 16);
  v16bf r;
#pragma unroll
  for (int j = 0; j < 8; ++j) { r[j] = lo[j]; r[j + 8] = hi[j]; }
  return r;
}

// ---------------------------------------------------------------------------
// Kernel 0: one-shot f32 -> bf16 conversion (x and the four weights).
// ---------------------------------------------------------------------------
__global__ __launch_bounds__(256)
void cvt_kernel(const float* __restrict__ src, __bf16* __restrict__ dst, int n4) {
  int i = blockIdx.x * 256 + threadIdx.x;
  if (i < n4) {
    float4 f = ((const float4*)src)[i];
    v4bf t; t[0] = f2bf(f.x); t[1] = f2bf(f.y); t[2] = f2bf(f.z); t[3] = f2bf(f.w);
    ((v4bf*)dst)[i] = t;
  }
}

// ---------------------------------------------------------------------------
// Kernel 1: Q/K/V projections, y = x @ W^T (W row-major [out][in] so staging
// W rows directly gives Bs[n][k]). Block 128(M) x 64(N), K-chunk 64,
// double-buffered async global->LDS; 8 waves x (2x2) 16x16 WMMA tiles.
// ---------------------------------------------------------------------------
__global__ __launch_bounds__(256)
void qkv_kernel(const __bf16* __restrict__ Xb, const __bf16* __restrict__ Wqb,
                const __bf16* __restrict__ Wkb, const __bf16* __restrict__ Wvb,
                __bf16* __restrict__ Qb, __bf16* __restrict__ Kb,
                __bf16* __restrict__ Vb) {
  const __bf16* W = (blockIdx.z == 0) ? Wqb : (blockIdx.z == 1) ? Wkb : Wvb;
  __bf16* Out     = (blockIdx.z == 0) ? Qb  : (blockIdx.z == 1) ? Kb  : Vb;
  const int m0 = blockIdx.x * 128, n0 = blockIdx.y * 64;

  __shared__ __bf16 As[2][128][72];   // +8 pad: rows 144B (16B aligned)
  __shared__ __bf16 Bs[2][64][72];

  const int tid = threadIdx.x, wid = tid >> 5, lane = tid & 31;
  const int l16 = lane & 15, lhi = lane >> 4;
  const int mwt = wid & 3, nwt = wid >> 2;

  auto load_tiles = [&](int buf, int k0) {
#pragma unroll
    for (int i = 0; i < 4; ++i) {           // A: 128x64 bf16 (4x16B/thread)
      int s = tid + i * 256; int m = s >> 3, d0 = (s & 7) * 8;
      copy16_g2l(Xb + (size_t)(m0 + m) * DB + k0 + d0, &As[buf][m][d0]);
    }
#pragma unroll
    for (int i = 0; i < 2; ++i) {           // B: 64x64 bf16 (2x16B/thread)
      int s = tid + i * 256; int nn = s >> 3, d0 = (s & 7) * 8;
      copy16_g2l(W + (size_t)(n0 + nn) * DB + k0 + d0, &Bs[buf][nn][d0]);
    }
  };

  v8f acc00 = {}, acc01 = {}, acc10 = {}, acc11 = {};
  load_tiles(0, 0);
  int buf = 0;
  for (int k0 = 0; k0 < DB; k0 += 64, buf ^= 1) {
    if (k0 + 64 < DB) { load_tiles(buf ^ 1, k0 + 64); async_wait6(); }
    else              { async_wait0(); }
    __syncthreads();                        // chunk k0 visible to all waves
#pragma unroll
    for (int ks = 0; ks < 64; ks += 32) {
      v16bf a0 = ldfragA(&As[buf][32 * mwt + l16][0], ks, lhi);
      v16bf a1 = ldfragA(&As[buf][32 * mwt + 16 + l16][0], ks, lhi);
      v16bf b0 = ldfragB(&Bs[buf][32 * nwt + l16][0], ks, lhi);
      v16bf b1 = ldfragB(&Bs[buf][32 * nwt + 16 + l16][0], ks, lhi);
      acc00 = wmma_bf16(a0, b0, acc00);
      acc01 = wmma_bf16(a0, b1, acc01);
      acc10 = wmma_bf16(a1, b0, acc10);
      acc11 = wmma_bf16(a1, b1, acc11);
    }
    __syncthreads();                        // done reading buf before reuse
  }

  const int rbase = m0 + 32 * mwt + 8 * lhi;   // D layout: M = r + 8*lhi
  const int cbase = n0 + 32 * nwt + l16;       // N = lane%16
#pragma unroll
  for (int r = 0; r < 8; ++r) {
    Out[(size_t)(rbase + r) * DB + cbase]           = f2bf(acc00[r]);
    Out[(size_t)(rbase + r) * DB + cbase + 16]      = f2bf(acc01[r]);
    Out[(size_t)(rbase + 16 + r) * DB + cbase]      = f2bf(acc10[r]);
    Out[(size_t)(rbase + 16 + r) * DB + cbase + 16] = f2bf(acc11[r]);
  }
}

// ---------------------------------------------------------------------------
// Kernel 2: denomInv[n,h] = masksum(n) ^ -sigmoid(norm_const[h])
// ---------------------------------------------------------------------------
__global__ __launch_bounds__(256)
void denom_kernel(const unsigned char* __restrict__ mask,
                  const float* __restrict__ norm_const,
                  float* __restrict__ denomInv) {
  __shared__ float red[256];
  const int n = blockIdx.x, tid = threadIdx.x;
  float s = 0.f;
  for (int i = tid; i < SB; i += 256) s += mask[n * SB + i] ? 1.0f : 0.0f;
  red[tid] = s;
  __syncthreads();
  for (int off = 128; off > 0; off >>= 1) {
    if (tid < off) red[tid] += red[tid + off];
    __syncthreads();
  }
  if (tid < HB) {
    float sig = 1.0f / (1.0f + expf(-norm_const[tid]));
    denomInv[n * HB + tid] = powf(red[0], -sig);
  }
}

// ---------------------------------------------------------------------------
// Kernel 3: per-(n,s,h) row of 64: q /= max(||q||,eps); k same * mask;
// v *= mask * denomInv. One wave32 per row, 2 elems/lane, shfl_xor reduce.
// ---------------------------------------------------------------------------
__global__ __launch_bounds__(256)
void normscale_kernel(__bf16* __restrict__ Qb, __bf16* __restrict__ Kb,
                      __bf16* __restrict__ Vb,
                      const unsigned char* __restrict__ mask,
                      const float* __restrict__ denomInv) {
  const int wid = threadIdx.x >> 5, lane = threadIdx.x & 31;
  const int row = blockIdx.x * 8 + wid;          // over N*S*H
  const int h = row & 15;
  const int ns = row >> 4;                       // n*S + s  (S = 2^12)
  const int n = ns >> 12;
  const size_t base = (size_t)ns * DB + h * HDB + lane * 2;

  const float m    = mask[ns] ? 1.0f : 0.0f;
  const float dinv = denomInv[(n << 4) + h];

  float q0 = bf2f(Qb[base]), q1 = bf2f(Qb[base + 1]);
  float k0 = bf2f(Kb[base]), k1 = bf2f(Kb[base + 1]);
  float ssq = q0 * q0 + q1 * q1;
  float ssk = k0 * k0 + k1 * k1;
#pragma unroll
  for (int off = 16; off > 0; off >>= 1) {
    ssq += __shfl_xor(ssq, off, 32);
    ssk += __shfl_xor(ssk, off, 32);
  }
  const float iq = 1.0f / fmaxf(sqrtf(ssq), 1e-12f);
  const float ik = m / fmaxf(sqrtf(ssk), 1e-12f);
  const float iv = m * dinv;

  Qb[base] = f2bf(q0 * iq); Qb[base + 1] = f2bf(q1 * iq);
  Kb[base] = f2bf(k0 * ik); Kb[base + 1] = f2bf(k1 * ik);
  float v0 = bf2f(Vb[base]), v1 = bf2f(Vb[base + 1]);
  Vb[base] = f2bf(v0 * iv); Vb[base + 1] = f2bf(v1 * iv);
}

// ---------------------------------------------------------------------------
// Kernel 4: kv[n,h] = k^T v : [64 x 4096] * [4096 x 64]. One block per (n,h).
// t-chunks of 64 staged transposed in LDS ([d][t], [e][t]) so fragment reads
// are contiguous. 8 waves: wave = (d-tile 0..3) x (e-pair 0..1).
// ---------------------------------------------------------------------------
__global__ __launch_bounds__(256)
void kv_kernel(const __bf16* __restrict__ Kb, const __bf16* __restrict__ Vb,
               float* __restrict__ kvbuf) {
  const int nh = blockIdx.x;
  const int n = nh >> 4, h = nh & 15;
  __shared__ __bf16 Kst[64][72];   // [d][t]
  __shared__ __bf16 Vst[64][72];   // [e][t]

  const int tid = threadIdx.x, wid = tid >> 5, lane = tid & 31;
  const int l16 = lane & 15, lhi = lane >> 4;
  const int dt = wid >> 1;
  const int ebase = (wid & 1) * 32;

  v8f acc0 = {}, acc1 = {};
  const size_t gbase = (size_t)n * SB * DB + h * HDB;

  for (int t0 = 0; t0 < SB; t0 += 64) {
#pragma unroll
    for (int it = 0; it < 2; ++it) {
      int s = tid + it * 256; int t = s >> 3, d0 = (s & 7) * 8;
      v8bf kk = *(const v8bf*)(Kb + gbase + (size_t)(t0 + t) * DB + d0);
      v8bf vv = *(const v8bf*)(Vb + gbase + (size_t)(t0 + t) * DB + d0);
#pragma unroll
      for (int j = 0; j < 8; ++j) { Kst[d0 + j][t] = kk[j]; Vst[d0 + j][t] = vv[j]; }
    }
    __syncthreads();
#pragma unroll
    for (int ks = 0; ks < 64; ks += 32) {
      v16bf a  = ldfragA(&Kst[dt * 16 + l16][0], ks, lhi);
      v16bf b0 = ldfragB(&Vst[ebase + l16][0], ks, lhi);
      v16bf b1 = ldfragB(&Vst[ebase + 16 + l16][0], ks, lhi);
      acc0 = wmma_bf16(a, b0, acc0);
      acc1 = wmma_bf16(a, b1, acc1);
    }
    __syncthreads();
  }

  float* out = kvbuf + (size_t)nh * HDB * HDB;
  const int drow = dt * 16 + 8 * lhi;
#pragma unroll
  for (int r = 0; r < 8; ++r) {
    out[(drow + r) * HDB + ebase + l16]      = acc0[r];
    out[(drow + r) * HDB + ebase + 16 + l16] = acc1[r];
  }
}

// ---------------------------------------------------------------------------
// Kernel 5: attn = q_hat @ kv per (n,h): [4096 x 64] * [64 x 64].
// Block: 128 s-rows x full head; writes attn in [n,s,D] layout (bf16).
// ---------------------------------------------------------------------------
__global__ __launch_bounds__(256)
void attn_kernel(const __bf16* __restrict__ Qb, const float* __restrict__ kvbuf,
                 __bf16* __restrict__ Attn) {
  const int stile = blockIdx.x, nh = blockIdx.y;
  const int n = nh >> 4, h = nh & 15;
  __shared__ __bf16 Qs[128][72];   // [s][d]
  __shared__ __bf16 KVs[64][72];   // [e][d]

  const int tid = threadIdx.x, wid = tid >> 5, lane = tid & 31;
  const int l16 = lane & 15, lhi = lane >> 4;
  const int mwt = wid & 3, nwt = wid >> 2;
  const size_t rowbase = (size_t)n * SB + stile * 128;

#pragma unroll
  for (int i = 0; i < 4; ++i) {                 // Q tile: async copy path
    int s = tid + i * 256; int m = s >> 3, d0 = (s & 7) * 8;
    copy16_g2l(Qb + (rowbase + m) * DB + h * HDB + d0, &Qs[m][d0]);
  }
#pragma unroll
  for (int i = 0; i < 4; ++i) {                 // kv f32 -> bf16, transposed
    int s = tid + i * 256; int d = s >> 4, quad = (s & 15) * 4;
    float4 f = *(const float4*)(kvbuf + (size_t)nh * HDB * HDB + d * HDB + quad);
    KVs[quad + 0][d] = f2bf(f.x);
    KVs[quad + 1][d] = f2bf(f.y);
    KVs[quad + 2][d] = f2bf(f.z);
    KVs[quad + 3][d] = f2bf(f.w);
  }
  async_wait0();
  __syncthreads();

  v8f acc00 = {}, acc01 = {}, acc10 = {}, acc11 = {};
#pragma unroll
  for (int ks = 0; ks < 64; ks += 32) {
    v16bf a0 = ldfragA(&Qs[32 * mwt + l16][0], ks, lhi);
    v16bf a1 = ldfragA(&Qs[32 * mwt + 16 + l16][0], ks, lhi);
    v16bf b0 = ldfragB(&KVs[32 * nwt + l16][0], ks, lhi);
    v16bf b1 = ldfragB(&KVs[32 * nwt + 16 + l16][0], ks, lhi);
    acc00 = wmma_bf16(a0, b0, acc00);
    acc01 = wmma_bf16(a0, b1, acc01);
    acc10 = wmma_bf16(a1, b0, acc10);
    acc11 = wmma_bf16(a1, b1, acc11);
  }

  const int rb = 32 * mwt + 8 * lhi;
  const int cb = h * HDB + 32 * nwt + l16;
#pragma unroll
  for (int r = 0; r < 8; ++r) {
    Attn[(rowbase + rb + r) * DB + cb]           = f2bf(acc00[r]);
    Attn[(rowbase + rb + r) * DB + cb + 16]      = f2bf(acc01[r]);
    Attn[(rowbase + rb + 16 + r) * DB + cb]      = f2bf(acc10[r]);
    Attn[(rowbase + rb + 16 + r) * DB + cb + 16] = f2bf(acc11[r]);
  }
}

// ---------------------------------------------------------------------------
// Kernel 6: out = attn @ Wo^T, f32 output. Same double-buffered async tiling.
// ---------------------------------------------------------------------------
__global__ __launch_bounds__(256)
void out_kernel(const __bf16* __restrict__ Attn, const __bf16* __restrict__ Wob,
                float* __restrict__ Out) {
  const int m0 = blockIdx.x * 128, n0 = blockIdx.y * 64;
  __shared__ __bf16 As[2][128][72];
  __shared__ __bf16 Bs[2][64][72];

  const int tid = threadIdx.x, wid = tid >> 5, lane = tid & 31;
  const int l16 = lane & 15, lhi = lane >> 4;
  const int mwt = wid & 3, nwt = wid >> 2;

  auto load_tiles = [&](int buf, int k0) {
#pragma unroll
    for (int i = 0; i < 4; ++i) {
      int s = tid + i * 256; int m = s >> 3, d0 = (s & 7) * 8;
      copy16_g2l(Attn + (size_t)(m0 + m) * DB + k0 + d0, &As[buf][m][d0]);
    }
#pragma unroll
    for (int i = 0; i < 2; ++i) {
      int s = tid + i * 256; int nn = s >> 3, d0 = (s & 7) * 8;
      copy16_g2l(Wob + (size_t)(n0 + nn) * DB + k0 + d0, &Bs[buf][nn][d0]);
    }
  };

  v8f acc00 = {}, acc01 = {}, acc10 = {}, acc11 = {};
  load_tiles(0, 0);
  int buf = 0;
  for (int k0 = 0; k0 < DB; k0 += 64, buf ^= 1) {
    if (k0 + 64 < DB) { load_tiles(buf ^ 1, k0 + 64); async_wait6(); }
    else              { async_wait0(); }
    __syncthreads();
#pragma unroll
    for (int ks = 0; ks < 64; ks += 32) {
      v16bf a0 = ldfragA(&As[buf][32 * mwt + l16][0], ks, lhi);
      v16bf a1 = ldfragA(&As[buf][32 * mwt + 16 + l16][0], ks, lhi);
      v16bf b0 = ldfragB(&Bs[buf][32 * nwt + l16][0], ks, lhi);
      v16bf b1 = ldfragB(&Bs[buf][32 * nwt + 16 + l16][0], ks, lhi);
      acc00 = wmma_bf16(a0, b0, acc00);
      acc01 = wmma_bf16(a0, b1, acc01);
      acc10 = wmma_bf16(a1, b0, acc10);
      acc11 = wmma_bf16(a1, b1, acc11);
    }
    __syncthreads();
  }

  const int rbase = m0 + 32 * mwt + 8 * lhi;
  const int cbase = n0 + 32 * nwt + l16;
#pragma unroll
  for (int r = 0; r < 8; ++r) {
    Out[(size_t)(rbase + r) * DB + cbase]           = acc00[r];
    Out[(size_t)(rbase + r) * DB + cbase + 16]      = acc01[r];
    Out[(size_t)(rbase + 16 + r) * DB + cbase]      = acc10[r];
    Out[(size_t)(rbase + 16 + r) * DB + cbase + 16] = acc11[r];
  }
}

// ---------------------------------------------------------------------------
extern "C" void kernel_launch(void* const* d_in, const int* in_sizes, int n_in,
                              void* d_out, int out_size, void* d_ws, size_t ws_size,
                              hipStream_t stream) {
  (void)in_sizes; (void)n_in; (void)out_size; (void)ws_size;
  const float*         x          = (const float*)d_in[0];
  const unsigned char* mask       = (const unsigned char*)d_in[1];  // numpy bool
  const float*         Wq         = (const float*)d_in[2];
  const float*         Wk         = (const float*)d_in[3];
  const float*         Wv         = (const float*)d_in[4];
  const float*         Wo         = (const float*)d_in[5];
  const float*         norm_const = (const float*)d_in[6];
  float*               out        = (float*)d_out;

  // Workspace layout (~169 MB):
  //   Xb (32M) | Wqb/Wkb/Wvb/Wob (4x2M) | Qb/Kb/Vb/Attn (4x32M) | kv (1M) | denomInv
  char* p = (char*)d_ws;
  const size_t NSD = (size_t)NB * SB * DB;
  const size_t DD  = (size_t)DB * DB;
  __bf16* Xb   = (__bf16*)p;  p += NSD * 2;
  __bf16* Wqb  = (__bf16*)p;  p += DD * 2;
  __bf16* Wkb  = (__bf16*)p;  p += DD * 2;
  __bf16* Wvb  = (__bf16*)p;  p += DD * 2;
  __bf16* Wob  = (__bf16*)p;  p += DD * 2;
  __bf16* Qb   = (__bf16*)p;  p += NSD * 2;
  __bf16* Kb   = (__bf16*)p;  p += NSD * 2;
  __bf16* Vb   = (__bf16*)p;  p += NSD * 2;
  __bf16* Attn = (__bf16*)p;  p += NSD * 2;
  float*  kvbuf    = (float*)p;  p += (size_t)NB * HB * HDB * HDB * 4;
  float*  denomInv = (float*)p;

  cvt_kernel<<<(int)(NSD / 4 / 256), 256, 0, stream>>>(x, Xb, (int)(NSD / 4));
  cvt_kernel<<<(int)(DD / 4 / 256), 256, 0, stream>>>(Wq, Wqb, (int)(DD / 4));
  cvt_kernel<<<(int)(DD / 4 / 256), 256, 0, stream>>>(Wk, Wkb, (int)(DD / 4));
  cvt_kernel<<<(int)(DD / 4 / 256), 256, 0, stream>>>(Wv, Wvb, (int)(DD / 4));
  cvt_kernel<<<(int)(DD / 4 / 256), 256, 0, stream>>>(Wo, Wob, (int)(DD / 4));

  qkv_kernel<<<dim3(SB * NB / 128, DB / 64, 3), 256, 0, stream>>>(
      Xb, Wqb, Wkb, Wvb, Qb, Kb, Vb);
  denom_kernel<<<NB, 256, 0, stream>>>(mask, norm_const, denomInv);
  normscale_kernel<<<NB * SB * HB / 8, 256, 0, stream>>>(Qb, Kb, Vb, mask, denomInv);
  kv_kernel<<<NB * HB, 256, 0, stream>>>(Kb, Vb, kvbuf);
  attn_kernel<<<dim3(SB / 128, NB * HB), 256, 0, stream>>>(Qb, kvbuf, Attn);
  out_kernel<<<dim3(SB * NB / 128, DB / 64), 256, 0, stream>>>(Attn, Wob, out);
}
